// HyperModulatedConv2d_42167988912485
// MI455X (gfx1250) — matrix operand order repaired and checked by reading, not compiled, not run on hardware
//
#include <hip/hip_runtime.h>
#include <hip/hip_bf16.h>
#include <stdint.h>

typedef __attribute__((ext_vector_type(16))) _Float16 v16h;
typedef __attribute__((ext_vector_type(8)))  _Float16 v8h;
typedef __attribute__((ext_vector_type(8)))  float    v8f;
typedef __attribute__((ext_vector_type(4)))  float    v4f;
typedef __attribute__((ext_vector_type(2)))  float    v2f;
typedef __attribute__((ext_vector_type(4)))  uint32_t u32x4;
typedef __attribute__((ext_vector_type(4)))  int      i32x4;
typedef __attribute__((ext_vector_type(8)))  int      i32x8;

#define BATCH   16
#define IN_C    128
#define OUT_C   128
#define HH      64
#define WW      64
#define ZDIM    256
#define KTAPS   9
#define KDIM    (IN_C * KTAPS)          // 1152
#define OUT_DIM (OUT_C * KDIM)          // 147456
#define ASTRIDE 40                      // padded LDS row stride for A (halves)
#define BSTRIDE 40                      // padded LDS row stride for B (halves)
#define NKSTEP  (KDIM / 32)             // 36

#if defined(__has_builtin)
# if __has_builtin(__builtin_amdgcn_tensor_load_to_lds)
#  define USE_TDM 1
# endif
#endif
#ifndef USE_TDM
# define USE_TDM 0
#endif

// ---------------------------------------------------------------------------
// Kernel 1: hypernetwork GEMM, fp32 WMMA (memory-bound: streams 151MB head_w)
//   D[o_tile(16) x b(16)] += head_w_tile(16x4) * z^T(4x16), K=256
//   w16[b][o_flat] = f16(base[o_flat] + head_b[o_flat] + D)
// ---------------------------------------------------------------------------
__global__ __launch_bounds__(256) void hyper_weights_kernel(
    const float* __restrict__ z,        // [16][256]
    const float* __restrict__ base_w,   // [OUT_DIM]
    const float* __restrict__ head_w,   // [OUT_DIM][256]
    const float* __restrict__ head_b,   // [OUT_DIM]
    _Float16*    __restrict__ w16)      // [16][OUT_DIM]
{
    const int lane = threadIdx.x & 31;
    const int wave = threadIdx.x >> 5;
    const int tile = blockIdx.x * 8 + wave;     // 9216 o-tiles total
    const int o0   = tile * 16;
    const int m    = lane & 15;
    const int koff = (lane < 16) ? 0 : 2;       // fp32 A: K split across lane halves

    const float* __restrict__ arow = head_w + (size_t)(o0 + m) * ZDIM + koff;
    const float* __restrict__ brow = z      + (size_t)m * ZDIM + koff;

    v8f acc = {};
#if __has_builtin(__builtin_amdgcn_wmma_f32_16x16x4_f32)
    #pragma unroll 4
    for (int k0 = 0; k0 < ZDIM; k0 += 4) {
        if ((k0 & 31) == 0)
            __builtin_prefetch(arow + k0 + 64, 0, 0);   // stream-ahead on head_w
        v2f a, b;
        a.x = arow[k0];  a.y = arow[k0 + 1];
        b.x = brow[k0];  b.y = brow[k0 + 1];
        acc = __builtin_amdgcn_wmma_f32_16x16x4_f32(
            /*neg_a=*/false, a, /*neg_b=*/false, b,
            /*c_mod=*/(short)0, acc, /*reuse_a=*/false, /*reuse_b=*/false);
    }
#else
    for (int k0 = 0; k0 < ZDIM; k0 += 4) {
        acc[0] += arow[k0] * brow[k0] + arow[k0 + 1] * brow[k0 + 1];
    }
#endif

    const int bcol  = lane & 15;                 // batch index (N)
    const int mbase = (lane < 16) ? 0 : 8;       // C/D layout: VGPR r -> M=r(+8)
    #pragma unroll
    for (int r = 0; r < 8; ++r) {
        const int of = o0 + mbase + r;
        const float v = acc[r] + head_b[of] + base_w[of];
        w16[(size_t)bcol * OUT_DIM + of] = (_Float16)v;
    }
}

// ---------------------------------------------------------------------------
// Kernel 2: x fp32 -> f16 (one pass, 33.5MB -> 16.8MB)
// ---------------------------------------------------------------------------
__global__ __launch_bounds__(256) void cvt_x_kernel(
    const float* __restrict__ x, _Float16* __restrict__ x16, int n)
{
    const int i = (blockIdx.x * blockDim.x + threadIdx.x) * 4;
    if (i + 3 < n) {
        v4f v = *(const v4f*)(x + i);
        _Float16* p = x16 + i;
        p[0] = (_Float16)v.x; p[1] = (_Float16)v.y;
        p[2] = (_Float16)v.z; p[3] = (_Float16)v.w;
    }
}

// ---------------------------------------------------------------------------
// TDM issue: load A tile (OUT_C rows x 32 halves, row stride KDIM halves)
// from w16 into LDS at byte offset lds_off, with HW padding to ASTRIDE=40
// halves per row (pad_interval=16 DWORDs, pad_amount=4 DWORDs).
// ---------------------------------------------------------------------------
#if USE_TDM
__device__ __forceinline__ void tdm_load_A(uint64_t gaddr, uint32_t lds_off)
{
    u32x4 g0;
    g0[0] = 1u;                                            // count=1 (valid)
    g0[1] = lds_off;                                       // lds_addr
    g0[2] = (uint32_t)gaddr;                               // global_addr[31:0]
    g0[3] = (uint32_t)((gaddr >> 32) & 0x01FFFFFFu)        // global_addr[56:32]
          | (2u << 30);                                    // type=2 (image)
    i32x8 g1;
    g1[0] = (1 << 16)        // data_size = 2 bytes
          | (1 << 20)        // pad_enable
          | (3 << 22)        // pad_interval: 16 DWORDs (one 64B row)
          | (3 << 25);       // pad_amount: 4 DWORDs (8 halves) -> 40-half stride
    g1[1] = (KDIM & 0xFFFF) << 16;                         // tensor_dim0[15:0]
    g1[2] = (KDIM >> 16) | ((OUT_C & 0xFFFF) << 16);       // dim0[31:16] | dim1[15:0]
    g1[3] = (OUT_C >> 16) | (32 << 16);                    // dim1[31:16] | tile_dim0=32
    g1[4] = OUT_C;                                         // tile_dim1=128, tile_dim2=0
    g1[5] = KDIM;                                          // tensor_dim0_stride[31:0]
    g1[6] = 0;                                             // stride hi | dim1_stride lo
    g1[7] = 0;
    i32x4 gz = {0, 0, 0, 0};
#if __clang_major__ >= 23
    i32x8 gz8 = {};
    __builtin_amdgcn_tensor_load_to_lds(g0, g1, gz, gz, gz8, 0);
#else
    __builtin_amdgcn_tensor_load_to_lds(g0, g1, gz, gz, 0);
#endif
}
#endif

// ---------------------------------------------------------------------------
// Kernel 3: per-sample conv as implicit GEMM on f16 WMMA, fp32 accumulate.
//   Block = (sample b, image row y): M=128 over 8 waves, N=64 pixels, K=1152.
//   A tile (128x32 halves) DMA'd to LDS by the TDM, double-buffered,
//   overlapped with compute via TENSORcnt. B = im2col tile staged by VALU.
// ---------------------------------------------------------------------------
__global__ __launch_bounds__(256) void hyperconv_kernel(
    const _Float16* __restrict__ x16,   // [16][128][64][64]
    const _Float16* __restrict__ w16,   // [16][OUT_C][KDIM]
    float*          __restrict__ out)   // [16][128][64][64]
{
    __shared__ __align__(16) _Float16 As[2][OUT_C * ASTRIDE];  // 2 x 10240 B
    __shared__ __align__(16) _Float16 Bs[WW * BSTRIDE];        // 5120 B

    const int b    = blockIdx.y;
    const int y    = blockIdx.x;
    const int tid  = threadIdx.x;
    const int lane = tid & 31;
    const int wave = tid >> 5;

    const int o0   = wave * 16;                  // this wave's M tile
    const int m    = lane & 15;
    const int aoff = (lane < 16) ? 0 : 8;        // f16 A layout: lane half picks K+8
    const int bko  = (lane < 16) ? 0 : 8;        // f16 B layout: lane half picks K+8

    const _Float16* __restrict__ xb = x16 + (size_t)b * IN_C * HH * WW;

#if USE_TDM
    const uint64_t wbase = (uint64_t)(uintptr_t)w16 + (uint64_t)b * OUT_DIM * 2;
    const uint32_t lds_a0 = (uint32_t)(uintptr_t)(&As[0][0]);
    const uint32_t lds_a1 = (uint32_t)(uintptr_t)(&As[1][0]);
    if (wave == 0) tdm_load_A(wbase, lds_a0);    // prologue: tile for k0=0
#else
    const _Float16* __restrict__ arow_g =
        w16 + (size_t)b * OUT_DIM + (size_t)(o0 + m) * KDIM + aoff;
#endif

    // staging task: each thread owns (k = tid/8, 8-pixel segment = tid%8)
    const int ks = tid >> 3;
    const int n0 = (tid & 7) * 8;

    v8f acc[4] = {{}, {}, {}, {}};

    for (int s = 0; s < NKSTEP; ++s) {
        const int k0 = s * 32;

        // ---- stage im2col B tile into LDS (overlaps TDM of A tile) ----
        {
            const int kg = k0 + ks;
            const int c  = kg / KTAPS;
            const int t  = kg - c * KTAPS;
            const int dy = t / 3 - 1;
            const int dx = t % 3 - 1;
            const int yy = y + dy;
            const bool yok = (yy >= 0) && (yy < HH);
            const _Float16* __restrict__ xrow = xb + ((size_t)c * HH + yy) * WW;
            #pragma unroll
            for (int i = 0; i < 8; ++i) {
                const int n  = n0 + i;
                const int xx = n + dx;
                _Float16 v = (_Float16)0.0f;
                if (yok && xx >= 0 && xx < WW) v = xrow[xx];
                Bs[n * BSTRIDE + ks] = v;
            }
        }
#if USE_TDM
        __builtin_amdgcn_s_wait_tensorcnt(0);    // A tile for this step landed
#endif
        __syncthreads();                         // As[s&1] + Bs visible to all
#if USE_TDM
        if (wave == 0 && s + 1 < NKSTEP)         // prefetch next A tile via TDM
            tdm_load_A(wbase + (uint64_t)(k0 + 32) * 2, (s & 1) ? lds_a0 : lds_a1);
#endif

        // ---- A fragment: 16 halves/lane (K 0..7 / 8..15 and +16 per half) ----
        v16h a;
        {
#if USE_TDM
            const _Float16* ap = &As[s & 1][(o0 + m) * ASTRIDE + aoff];
            v8h lo = *(const v8h*)(ap);
            v8h hi = *(const v8h*)(ap + 16);
#else
            v8h lo = *(const v8h*)(arow_g + k0);
            v8h hi = *(const v8h*)(arow_g + k0 + 16);
#endif
            #pragma unroll
            for (int i = 0; i < 8; ++i) { a[i] = lo[i]; a[8 + i] = hi[i]; }
        }

        // ---- 4 N sub-tiles share the staged B tile ----
        #pragma unroll
        for (int j = 0; j < 4; ++j) {
            const _Float16* bp = &Bs[(j * 16 + m) * BSTRIDE + bko];
            v8h b0 = *(const v8h*)(bp);
            v8h b1 = *(const v8h*)(bp + 16);
            v16h bb;
            #pragma unroll
            for (int i = 0; i < 8; ++i) { bb[i] = b0[i]; bb[8 + i] = b1[i]; }
            acc[j] = __builtin_amdgcn_wmma_f32_16x16x32_f16(
                /*neg_a=*/false, a, /*neg_b=*/false, bb,
                /*c_mod=*/(short)0, acc[j], /*reuse_a=*/false, /*reuse_b=*/false);
        }
        __syncthreads();                         // protect Bs before next stage
    }

    // ---- write D tiles: VGPR r -> M = r + (lane<16 ? 0 : 8), N = lane%16 ----
    const int mbase = (lane < 16) ? 0 : 8;
    #pragma unroll
    for (int j = 0; j < 4; ++j) {
        const int ncol = j * 16 + m;
        #pragma unroll
        for (int r = 0; r < 8; ++r) {
            const int o = o0 + mbase + r;
            out[(((size_t)b * OUT_C + o) * HH + y) * WW + ncol] = acc[j][r];
        }
    }
}

// ---------------------------------------------------------------------------
extern "C" void kernel_launch(void* const* d_in, const int* in_sizes, int n_in,
                              void* d_out, int out_size, void* d_ws, size_t ws_size,
                              hipStream_t stream)
{
    (void)in_sizes; (void)n_in; (void)out_size; (void)ws_size;

    const float* x      = (const float*)d_in[0];   // [16,128,64,64]
    const float* z      = (const float*)d_in[1];   // [16,256]
    const float* base_w = (const float*)d_in[2];   // [128,128,3,3] = [OUT_DIM]
    const float* head_w = (const float*)d_in[3];   // [OUT_DIM,256]
    const float* head_b = (const float*)d_in[4];   // [OUT_DIM]
    float*       out    = (float*)d_out;

    _Float16* w16 = (_Float16*)d_ws;                                   // 16*OUT_DIM halves
    _Float16* x16 = (_Float16*)((char*)d_ws +
                    (size_t)BATCH * OUT_DIM * sizeof(_Float16));       // 16*128*64*64 halves

    // 1) modulated weights: 9216 o-tiles / 8 waves per block
    hyper_weights_kernel<<<9216 / 8, 256, 0, stream>>>(z, base_w, head_w, head_b, w16);

    // 2) x -> f16
    const int nx = BATCH * IN_C * HH * WW;          // 8,388,608
    cvt_x_kernel<<<nx / 1024, 256, 0, stream>>>(x, x16, nx);

    // 3) conv: one block per (image row, sample)
    hyperconv_kernel<<<dim3(HH, BATCH), 256, 0, stream>>>(x16, w16, out);
}
